// DREAMAcousticNL_67388036874775
// MI455X (gfx1250) — compile-verified, matrix operand order checked by reading, not run on hardware
//
#include <hip/hip_runtime.h>
#include <hip/hip_bf16.h>
#include <cstdint>
#include <cstddef>

#define TLEN 1000
#define BSZ  16
#define MEL  80
#define MELP 96      // MEL padded to 3 K-chunks of 32
#define HID  512
#define NCLS 64
#define NWG  16      // persistent workgroups for the scan; NWG*32 == HID

typedef __attribute__((ext_vector_type(16))) __bf16 v16bf;
typedef __attribute__((ext_vector_type(8)))  float  v8f;

static __device__ inline v8f vzero8() {
  v8f z;
#pragma unroll
  for (int i = 0; i < 8; ++i) z[i] = 0.f;
  return z;
}

static __device__ inline float sigmoidf_(float x) { return 1.f / (1.f + __expf(-x)); }

// pack 4 fp32 -> 4 bf16, single 8-byte store (cuts ds ops 4x vs scalar b16 stores)
static __device__ inline void bf16x4_store(__bf16* dst, float a, float b, float c, float d) {
  union { __bf16 h[4]; unsigned long long u; } p;
  p.h[0] = (__bf16)a; p.h[1] = (__bf16)b; p.h[2] = (__bf16)c; p.h[3] = (__bf16)d;
  *reinterpret_cast<unsigned long long*>(dst) = p.u;
}

// 16-bit A/B fragment (16x32 tile) per CDNA5 ISA 7.12.2:
// lane L: row = L&15, half = L>>4; elem e<8 -> K = half*8+e ; e>=8 -> K = 16+half*8+(e-8)
static __device__ inline v16bf load_frag(const __bf16* __restrict__ base, int stride,
                                         int row, int k0, int half) {
  const __bf16* p = base + row * stride + k0 + half * 8;
  v16bf f;
#pragma unroll
  for (int e = 0; e < 16; ++e) f[e] = p[(e < 8) ? e : (8 + e)];
  return f;
}

static __device__ inline v8f wmma_bf16(v16bf a, v16bf b, v8f c) {
  return __builtin_amdgcn_wmma_f32_16x16x32_bf16(false, a, false, b, (short)0, c, false, false);
}

// ---------------------------------------------------------------------------
// fp32 -> bf16 weight conversion with optional column padding (rows x scols -> rows x dcols)
__global__ __launch_bounds__(256) void convert_pad_kernel(const float* __restrict__ src,
                                                          __bf16* __restrict__ dst,
                                                          int rows, int scols, int dcols) {
  int n = rows * dcols;
  for (int i = blockIdx.x * 256 + threadIdx.x; i < n; i += gridDim.x * 256) {
    int r = i / dcols, c = i % dcols;
    dst[i] = (c < scols) ? (__bf16)src[(size_t)r * scols + c] : (__bf16)0.f;
  }
}

// ---------------------------------------------------------------------------
// be0[t] = clip(x/||x||) @ B0^T    (parallel over t; M=16, N=512, K=80->96)
__global__ __launch_bounds__(256) void l0_base_kernel(const float* __restrict__ feats,
                                                      const __bf16* __restrict__ b0h, // [HID][MELP]
                                                      float* __restrict__ be0f) {
  const int t = blockIdx.x;
  __shared__ __align__(16) float xraw[BSZ * MEL];
  __shared__ float npart[BSZ * 16];
  __shared__ float snorm[BSZ];
  __shared__ __align__(16) __bf16 xs[BSZ * MELP];
  const int tid = threadIdx.x;

  for (int i = tid * 4; i < BSZ * MEL; i += 1024) {  // 80 % 4 == 0: chunks stay in-row
    int m = i / MEL, k = i % MEL;
    *(float4*)&xraw[i] = *(const float4*)&feats[((size_t)m * TLEN + t) * MEL + k];
  }
  __syncthreads();
  { // parallel row norms: 16 chunks of 5 per row, fixed-order reduce
    int m = tid >> 4, c = tid & 15;
    float s = 0.f;
    for (int k = c * 5; k < c * 5 + 5; ++k) { float v = xraw[m * MEL + k]; s += v * v; }
    npart[tid] = s;
  }
  __syncthreads();
  if (tid < BSZ) {
    float s = 0.f;
    for (int j = 0; j < 16; ++j) s += npart[tid * 16 + j];
    snorm[tid] = fmaxf(sqrtf(s), 1e-6f);
  }
  __syncthreads();
  for (int i = tid * 4; i < BSZ * MELP; i += 1024) {  // 96 % 4 == 0, 80 % 4 == 0
    int m = i / MELP, k = i % MELP;
    float inv = 1.f / snorm[m];
    float v[4];
#pragma unroll
    for (int j = 0; j < 4; ++j) {
      float x = (k + j < MEL) ? fminf(fmaxf(xraw[m * MEL + k + j] * inv, -1.f), 1.f) : 0.f;
      v[j] = x;
    }
    bf16x4_store(&xs[i], v[0], v[1], v[2], v[3]);
  }
  __syncthreads();

  const int lane = tid & 31, wv = tid >> 5, half = lane >> 4;
  for (int tile = wv; tile < HID / 16; tile += 8) {
    v8f acc = vzero8();
#pragma unroll
    for (int kc = 0; kc < MELP / 32; ++kc) {
      v16bf a = load_frag(xs, MELP, lane & 15, kc * 32, half);
      v16bf b = load_frag(b0h, MELP, tile * 16 + (lane & 15), kc * 32, half);
      acc = wmma_bf16(a, b, acc);
    }
    int nc = tile * 16 + (lane & 15);
#pragma unroll
    for (int r = 0; r < 8; ++r) {
      int m = r + 8 * half;
      be0f[((size_t)t * BSZ + m) * HID + nc] = acc[r];
    }
  }
}

// ---------------------------------------------------------------------------
// scale1[t] = max(||be0||,1e-6); be1[t] = clip(be0/scale1) @ B1^T   (parallel over t)
__global__ __launch_bounds__(256) void l1_base_kernel(const float* __restrict__ be0f,
                                                      const __bf16* __restrict__ b1h, // [HID][HID]
                                                      float* __restrict__ be1f,
                                                      float* __restrict__ scale1) {
  const int t = blockIdx.x;
  __shared__ __align__(16) float xr[BSZ * HID];
  __shared__ float npart[BSZ * 16];
  __shared__ float sn[BSZ];
  __shared__ __align__(16) __bf16 xs[BSZ * HID];
  const int tid = threadIdx.x;

  for (int i = tid * 4; i < BSZ * HID; i += 1024)
    *(float4*)&xr[i] = *(const float4*)&be0f[(size_t)t * BSZ * HID + i];
  __syncthreads();
  { // parallel row norms: 16 chunks of 32 per row
    int m = tid >> 4, c = tid & 15;
    float s = 0.f;
    for (int k = c * 32; k < c * 32 + 32; ++k) { float v = xr[m * HID + k]; s += v * v; }
    npart[tid] = s;
  }
  __syncthreads();
  if (tid < BSZ) {
    float s = 0.f;
    for (int j = 0; j < 16; ++j) s += npart[tid * 16 + j];
    s = fmaxf(sqrtf(s), 1e-6f);
    sn[tid] = s;
    scale1[t * BSZ + tid] = s;
  }
  __syncthreads();
  for (int i = tid * 4; i < BSZ * HID; i += 1024) {
    int m = i >> 9;
    float inv = 1.f / sn[m];
    float4 v = *(const float4*)&xr[i];
    bf16x4_store(&xs[i],
                 fminf(fmaxf(v.x * inv, -1.f), 1.f), fminf(fmaxf(v.y * inv, -1.f), 1.f),
                 fminf(fmaxf(v.z * inv, -1.f), 1.f), fminf(fmaxf(v.w * inv, -1.f), 1.f));
  }
  __syncthreads();

  const int lane = tid & 31, wv = tid >> 5, half = lane >> 4;
  for (int tile = wv; tile < HID / 16; tile += 8) {
    v8f acc = vzero8();
#pragma unroll 4
    for (int kc = 0; kc < HID / 32; ++kc) {
      v16bf a = load_frag(xs, HID, lane & 15, kc * 32, half);
      v16bf b = load_frag(b1h, HID, tile * 16 + (lane & 15), kc * 32, half);
      acc = wmma_bf16(a, b, acc);
    }
    int nc = tile * 16 + (lane & 15);
#pragma unroll
    for (int r = 0; r < 8; ++r) {
      int m = r + 8 * half;
      be1f[((size_t)t * BSZ + m) * HID + nc] = acc[r];
    }
  }
}

// ---------------------------------------------------------------------------
// two-phase-per-step grid barrier: CDNA5 cluster barrier fast path (NOP when not
// dispatched as a cluster, per ISA 3.2) + monotonic-phase atomic barrier.
static __device__ inline void grid_barrier(unsigned* __restrict__ slots, unsigned phase) {
  __syncthreads();
  __builtin_amdgcn_s_cluster_barrier();
  if (threadIdx.x == 0) {
    __threadfence();
    __atomic_store_n(&slots[blockIdx.x], phase, __ATOMIC_RELEASE);
  }
  if (threadIdx.x < NWG) {
    while (__atomic_load_n(&slots[threadIdx.x], __ATOMIC_ACQUIRE) < phase) {
      __builtin_amdgcn_s_sleep(2);
    }
  }
  __threadfence();
  __syncthreads();
}

// ---------------------------------------------------------------------------
// Persistent h1 scan. 16 WGs x 8 waves; WG `wg` owns columns [wg*32, wg*32+32).
// C1/W1 bf16 slices are staged into LDS with CDNA5 async LDS loads (ASYNCcnt)
// and stay resident for all 1000 steps. Per step:
//   A: p = h1 @ C1^T (slice) -> error slice + ||error||^2 partial -> barrier
//   B: surprise -> q = error @ W1^T (slice) -> h1 update (slice)   -> barrier
__global__ __launch_bounds__(256, 1) void scan_kernel(
    const float* __restrict__ be0f, const float* __restrict__ be1f,
    const float* __restrict__ scale1,
    const __bf16* __restrict__ c1h, const __bf16* __restrict__ w1h,
    const float* __restrict__ a1, const float* __restrict__ tau1p,
    const float* __restrict__ gam1p,
    float* __restrict__ h1cur, float* __restrict__ h1all,
    float* __restrict__ errbuf, float* __restrict__ errpart,
    unsigned* __restrict__ slots) {
  extern __shared__ __align__(16) char smem[];
  __bf16* c1s   = (__bf16*)smem;               // 32 x HID  (32 KB)
  __bf16* w1s   = c1s + 32 * HID;              // 32 x HID  (32 KB)
  __bf16* afrag = w1s + 32 * HID;              // 16 x HID  (16 KB)
  float*  part  = (float*)(afrag + BSZ * HID); // 8 waves x 256 (8 KB)
  float*  esq   = part + 8 * 256;              // 512
  float*  ssur  = esq + 512;                   // 16
  float*  ssc   = ssur + 16;                   // 16

  const int tid = threadIdx.x;
  const int wg = blockIdx.x;
  const int lane = tid & 31, wv = tid >> 5, half = lane >> 4;
  const int n0 = wg * 32;
  const int tile = wv & 1;      // which 16-col tile of our 32-col slice
  const int kpart = wv >> 1;    // which quarter of K (4 chunks of 32)

  { // stage resident weight slices (rows n0..n0+31) with async global->LDS b128
    // copies (ASYNCcnt-tracked, no VGPR round trip). LDS byte address = low 32
    // bits of the generic pointer (ISA 10.2).
    const unsigned ldsc = (unsigned)(uintptr_t)c1s;
    const unsigned ldsw = (unsigned)(uintptr_t)w1s;
    const unsigned long long gc = (unsigned long long)(uintptr_t)(c1h + (size_t)n0 * HID);
    const unsigned long long gw = (unsigned long long)(uintptr_t)(w1h + (size_t)n0 * HID);
    for (int i = tid; i < 32 * HID * 2 / 16; i += 256) {  // 2048 x 16B per matrix
      unsigned lc = ldsc + (unsigned)i * 16u;
      unsigned long long ac = gc + (unsigned long long)i * 16ull;
      asm volatile("global_load_async_to_lds_b128 %0, %1, off"
                   :: "v"(lc), "v"(ac) : "memory");
      unsigned lw = ldsw + (unsigned)i * 16u;
      unsigned long long aw = gw + (unsigned long long)i * 16ull;
      asm volatile("global_load_async_to_lds_b128 %0, %1, off"
                   :: "v"(lw), "v"(aw) : "memory");
    }
    asm volatile("s_wait_asynccnt 0x0" ::: "memory");
  }
  const float tau = tau1p[0], gam = gam1p[0];
  unsigned phase = 0;
  __syncthreads();

  for (int t = 0; t < TLEN; ++t) {
    // ---------------- phase A ----------------
    for (int i = tid * 4; i < BSZ * HID; i += 1024) {
      float4 v = *(const float4*)&h1cur[i];
      bf16x4_store(&afrag[i], v.x, v.y, v.z, v.w);
    }
    if (tid < BSZ) ssc[tid] = scale1[t * BSZ + tid];
    __syncthreads();

    v8f acc = vzero8();
#pragma unroll
    for (int kc = 0; kc < 4; ++kc) {
      int k0 = (kpart * 4 + kc) * 32;
      v16bf a = load_frag(afrag, HID, lane & 15, k0, half);
      v16bf b = load_frag(c1s, HID, tile * 16 + (lane & 15), k0, half);
      acc = wmma_bf16(a, b, acc);
    }
#pragma unroll
    for (int r = 0; r < 8; ++r) part[wv * 256 + r * 32 + lane] = acc[r];
    __syncthreads();

    for (int o = tid; o < 512; o += 256) {
      int tl = o >> 8, r = (o >> 5) & 7, ln = o & 31;
      float s = 0.f;
#pragma unroll
      for (int kp = 0; kp < 4; ++kp) s += part[(kp * 2 + tl) * 256 + r * 32 + ln];
      int m = r + 8 * (ln >> 4);
      int gcol = n0 + tl * 16 + (ln & 15);
      float e = be0f[((size_t)t * BSZ + m) * HID + gcol] - tanhf(s) * ssc[m];
      errbuf[m * HID + gcol] = e;
      esq[o] = e * e;
    }
    __syncthreads();
    if (tid < BSZ) { // fixed-order (deterministic) partial row reduction
      int m = tid, r = m & 7, lh = (m >> 3) * 16;
      float s = 0.f;
      for (int tl = 0; tl < 2; ++tl)
        for (int c = 0; c < 16; ++c) s += esq[tl * 256 + r * 32 + lh + c];
      errpart[wg * BSZ + m] = s;
    }
    if (t + 1 < TLEN) { // pull next step's activations toward the WGP
      __builtin_prefetch((const void*)&be0f[((size_t)(t + 1) * BSZ) * HID + n0], 0, 1);
      __builtin_prefetch((const void*)&be1f[((size_t)(t + 1) * BSZ) * HID + n0], 0, 1);
    }
    grid_barrier(slots, ++phase);

    // ---------------- phase B ----------------
    if (tid < BSZ) {
      int m = tid;
      float tot = 0.f;
      for (int w = 0; w < NWG; ++w) tot += errpart[w * BSZ + m];
      float rel = fminf(sqrtf(tot) / ssc[m], 4.0f);
      ssur[m] = sigmoidf_((rel - tau) / gam);
    }
    for (int i = tid * 4; i < BSZ * HID; i += 1024) {
      float4 v = *(const float4*)&errbuf[i];
      bf16x4_store(&afrag[i], v.x, v.y, v.z, v.w);
    }
    __syncthreads();

    acc = vzero8();
#pragma unroll
    for (int kc = 0; kc < 4; ++kc) {
      int k0 = (kpart * 4 + kc) * 32;
      v16bf a = load_frag(afrag, HID, lane & 15, k0, half);
      v16bf b = load_frag(w1s, HID, tile * 16 + (lane & 15), k0, half);
      acc = wmma_bf16(a, b, acc);
    }
#pragma unroll
    for (int r = 0; r < 8; ++r) part[wv * 256 + r * 32 + lane] = acc[r];
    __syncthreads();

    for (int o = tid; o < 512; o += 256) {
      int tl = o >> 8, r = (o >> 5) & 7, ln = o & 31;
      float q = 0.f;
#pragma unroll
      for (int kp = 0; kp < 4; ++kp) q += part[(kp * 2 + tl) * 256 + r * 32 + ln];
      int m = r + 8 * (ln >> 4);
      int gcol = n0 + tl * 16 + (ln & 15);
      float h = h1cur[m * HID + gcol];
      float sur = ssur[m];
      float bev = be1f[((size_t)t * BSZ + m) * HID + gcol];
      float input_h = h * 0.2f + bev * 0.6f + q * sur * 0.2f;
      float g = sur * sigmoidf_(a1[gcol]);
      float hn = h * (1.f - g) + tanhf(input_h) * g;
      h1cur[m * HID + gcol] = hn;
      h1all[((size_t)t * BSZ + m) * HID + gcol] = hn;
    }
    grid_barrier(slots, ++phase);
  }
}

// ---------------------------------------------------------------------------
// out[b,t,:] = concat(h1[t,b,:], be1[t,b,:]) @ head_w^T + head_b (parallel over t)
__global__ __launch_bounds__(128) void head_kernel(const float* __restrict__ h1all,
                                                   const float* __restrict__ be1f,
                                                   const __bf16* __restrict__ hwh, // [NCLS][2*HID]
                                                   const float* __restrict__ head_b,
                                                   float* __restrict__ out) {
  const int t = blockIdx.x;
  __shared__ __align__(16) __bf16 xs[BSZ * 2 * HID];
  const int tid = threadIdx.x;
  for (int i = tid * 4; i < BSZ * 2 * HID; i += 512) {  // 1024 % 4 == 0; split at 512 is aligned
    int m = i >> 10, k = i & 1023;
    float4 v = (k < HID) ? *(const float4*)&h1all[((size_t)t * BSZ + m) * HID + k]
                         : *(const float4*)&be1f[((size_t)t * BSZ + m) * HID + (k - HID)];
    bf16x4_store(&xs[i], v.x, v.y, v.z, v.w);
  }
  __syncthreads();

  const int lane = tid & 31, wv = tid >> 5, half = lane >> 4;
  const int tile = wv; // NCLS/16 == 4 tiles == 4 waves
  v8f acc = vzero8();
#pragma unroll 4
  for (int kc = 0; kc < 2 * HID / 32; ++kc) {
    v16bf a = load_frag(xs, 2 * HID, lane & 15, kc * 32, half);
    v16bf b = load_frag(hwh, 2 * HID, tile * 16 + (lane & 15), kc * 32, half);
    acc = wmma_bf16(a, b, acc);
  }
  int col = tile * 16 + (lane & 15);
  float bias = head_b[col];
#pragma unroll
  for (int r = 0; r < 8; ++r) {
    int m = r + 8 * half;
    out[((size_t)m * TLEN + t) * NCLS + col] = acc[r] + bias;
  }
}

// ---------------------------------------------------------------------------
extern "C" void kernel_launch(void* const* d_in, const int* in_sizes, int n_in,
                              void* d_out, int out_size, void* d_ws, size_t ws_size,
                              hipStream_t stream) {
  (void)in_sizes; (void)n_in; (void)out_size; (void)ws_size;
  const float* feats  = (const float*)d_in[0];
  const float* B0     = (const float*)d_in[2];
  const float* C1     = (const float*)d_in[7];
  const float* B1     = (const float*)d_in[8];
  const float* W1     = (const float*)d_in[9];
  const float* a1     = (const float*)d_in[10];
  const float* tau01  = (const float*)d_in[11];
  const float* gamma1 = (const float*)d_in[12];
  const float* head_w = (const float*)d_in[13];
  const float* head_b = (const float*)d_in[14];
  float* out = (float*)d_out;

  char* ws = (char*)d_ws;
  size_t off = 0;
  auto take = [&](size_t bytes) { off = (off + 255) & ~(size_t)255; size_t o = off; off += bytes; return o; };
  __bf16* b0h  = (__bf16*)(ws + take((size_t)HID * MELP * 2));
  __bf16* b1h  = (__bf16*)(ws + take((size_t)HID * HID * 2));
  __bf16* c1h  = (__bf16*)(ws + take((size_t)HID * HID * 2));
  __bf16* w1h  = (__bf16*)(ws + take((size_t)HID * HID * 2));
  __bf16* hwh  = (__bf16*)(ws + take((size_t)NCLS * 2 * HID * 2));
  float* be0f    = (float*)(ws + take((size_t)TLEN * BSZ * HID * 4));
  float* be1f    = (float*)(ws + take((size_t)TLEN * BSZ * HID * 4));
  float* h1all   = (float*)(ws + take((size_t)TLEN * BSZ * HID * 4));
  float* scale1  = (float*)(ws + take((size_t)TLEN * BSZ * 4));
  float* h1cur   = (float*)(ws + take((size_t)BSZ * HID * 4));
  float* errbuf  = (float*)(ws + take((size_t)BSZ * HID * 4));
  float* errpart = (float*)(ws + take((size_t)NWG * BSZ * 4));
  unsigned* slots = (unsigned*)(ws + take((size_t)NWG * 4));

  hipMemsetAsync((void*)h1cur, 0, (size_t)BSZ * HID * 4, stream);
  hipMemsetAsync((void*)slots, 0, (size_t)NWG * 4, stream);

  convert_pad_kernel<<<96, 256, 0, stream>>>(B0, b0h, HID, MEL, MELP);
  convert_pad_kernel<<<256, 256, 0, stream>>>(B1, b1h, HID, HID, HID);
  convert_pad_kernel<<<256, 256, 0, stream>>>(C1, c1h, HID, HID, HID);
  convert_pad_kernel<<<256, 256, 0, stream>>>(W1, w1h, HID, HID, HID);
  convert_pad_kernel<<<128, 256, 0, stream>>>(head_w, hwh, NCLS, 2 * HID, 2 * HID);

  l0_base_kernel<<<TLEN, 256, 0, stream>>>(feats, b0h, be0f);
  l1_base_kernel<<<TLEN, 256, 0, stream>>>(be0f, b1h, be1f, scale1);

  const size_t scan_smem = (size_t)(32 * HID + 32 * HID + BSZ * HID) * 2 +
                           (8 * 256 + 512 + 32) * 4;
  scan_kernel<<<NWG, 256, scan_smem, stream>>>(be0f, be1f, scale1, c1h, w1h, a1,
                                               tau01, gamma1, h1cur, h1all, errbuf,
                                               errpart, slots);
  head_kernel<<<TLEN, 128, 0, stream>>>(h1all, be1f, hwh, head_b, out);
}
